// Decoder_80015240724649
// MI455X (gfx1250) — matrix operand order repaired and checked by reading, not compiled, not run on hardware
//
#include <hip/hip_runtime.h>
#include <hip/hip_fp16.h>

// Problem constants (from reference): B=128, N=256, D=128, H=8, KD=16
#define B_  128
#define N_  256
#define D_  128
#define H_  8
#define KD_ 16
#define C3_ 384   // 3*D context width

typedef _Float16 v16h __attribute__((ext_vector_type(16)));
typedef _Float16 h8   __attribute__((ext_vector_type(8)));
typedef float    v8f  __attribute__((ext_vector_type(8)));

// ---------------------------------------------------------------------------
// Generic f16 WMMA GEMM: C[M,N] = A[M,K] * Bt[N,K]^T  (Bt is packed [n][k])
// One wave per 16x16 C tile, 8 waves (tiles) per 256-thread block.
// A/B lane layouts: contiguous k = hi*16 + j in both (16-bit A 16x32, B 32x16).
// ---------------------------------------------------------------------------
__global__ __launch_bounds__(256) void gemm_f16(const _Float16* __restrict__ A,
                                                const _Float16* __restrict__ Bt,
                                                float* __restrict__ Cf,
                                                _Float16* __restrict__ Ch,
                                                int M, int Nn, int K) {
    const int lane = threadIdx.x & 31;
    const int wave = threadIdx.x >> 5;
    const int l15  = lane & 15;
    const int hi   = lane >> 4;              // half-wave select
    const int ntiles = Nn >> 4;
    const int tile = blockIdx.x * 8 + wave;
    const int mt = tile / ntiles;
    const int nt = tile - mt * ntiles;

    const _Float16* __restrict__ arow = A  + (size_t)(mt * 16 + l15) * K;
    const _Float16* __restrict__ bcol = Bt + (size_t)(nt * 16 + l15) * K;

    v8f acc = {};
    for (int k0 = 0; k0 < K; k0 += 32) {
        __builtin_prefetch(arow + k0 + 64, 0, 0);   // global_prefetch_b8
        __builtin_prefetch(bcol + k0 + 64, 0, 0);
        // A: elem j -> K = k0 + hi*16 + j  (contiguous, matches B)
        h8 alo = *(const h8*)(arow + k0 + hi * 16);
        h8 ahi = *(const h8*)(arow + k0 + hi * 16 + 8);
        v16h a = __builtin_shufflevector(alo, ahi,
                                         0,1,2,3,4,5,6,7,8,9,10,11,12,13,14,15);
        // B: elem j -> K = k0 + hi*16 + j  (contiguous since Bt is [n][k])
        v16h b = *(const v16h*)(bcol + k0 + hi * 16);
        acc = __builtin_amdgcn_wmma_f32_16x16x32_f16(false, a, false, b,
                                                     (short)0, acc, false, false);
    }
    const int ncol = nt * 16 + l15;
#pragma unroll
    for (int r = 0; r < 8; ++r) {            // VGPR r -> M = hi*8 + r
        int m = mt * 16 + hi * 8 + r;
        if (Cf) Cf[(size_t)m * Nn + ncol] = acc[r];
        else    Ch[(size_t)m * Nn + ncol] = (_Float16)acc[r];
    }
}

// ---------------------------------------------------------------------------
// Context projection GEMM (per step): ctx[b,e] = concat(ge,last,first)[b,:] @ Wc^T
// A (128x384) is materialized on the fly from the three f32 state vectors.
// ---------------------------------------------------------------------------
__global__ __launch_bounds__(256) void ctx_gemm(const float* __restrict__ ge,
                                                const float* __restrict__ lastv,
                                                const float* __restrict__ firstv,
                                                const _Float16* __restrict__ wc,
                                                _Float16* __restrict__ ctx) {
    const int lane = threadIdx.x & 31;
    const int wave = threadIdx.x >> 5;
    const int l15  = lane & 15;
    const int hi   = lane >> 4;
    const int tile = blockIdx.x * 8 + wave;  // 64 tiles total
    const int mt = tile >> 3;
    const int nt = tile & 7;
    const int m  = mt * 16 + l15;            // batch row
    const _Float16* __restrict__ bcol = wc + (size_t)(nt * 16 + l15) * C3_;

    v8f acc = {};
    for (int k0 = 0; k0 < C3_; k0 += 32) {   // segment boundaries align with 32
        const float* src = (k0 < 128) ? (ge     + m * D_ + k0)
                         : (k0 < 256) ? (lastv  + m * D_ + (k0 - 128))
                                      : (firstv + m * D_ + (k0 - 256));
        v16h a;
#pragma unroll
        for (int j = 0; j < 8; ++j) a[j]     = (_Float16)src[hi * 16 + j];
#pragma unroll
        for (int j = 0; j < 8; ++j) a[8 + j] = (_Float16)src[hi * 16 + 8 + j];
        v16h b = *(const v16h*)(bcol + k0 + hi * 16);
        acc = __builtin_amdgcn_wmma_f32_16x16x32_f16(false, a, false, b,
                                                     (short)0, acc, false, false);
    }
    const int ncol = nt * 16 + l15;
#pragma unroll
    for (int r = 0; r < 8; ++r) {
        int mm = mt * 16 + hi * 8 + r;
        ctx[(size_t)mm * D_ + ncol] = (_Float16)acc[r];
    }
}

// ---------------------------------------------------------------------------
// Attention core: per (h,b) pair (one wave each): compat -> softmax -> heads.
// These are batched GEMVs with per-b operands (not GEMMs) -> VALU + shuffles.
// ---------------------------------------------------------------------------
__global__ __launch_bounds__(256) void attn_k(const float* __restrict__ qall,
                                              const _Float16* __restrict__ khh,
                                              const _Float16* __restrict__ vhh,
                                              const int* __restrict__ mask,
                                              _Float16* __restrict__ heads) {
    const int lane = threadIdx.x & 31;
    const int wave = threadIdx.x >> 5;
    const int pair = blockIdx.x * 8 + wave;  // H_*B_ = 1024 pairs
    const int h = pair & (H_ - 1);
    const int b = pair >> 3;

    float qv[KD_];
#pragma unroll
    for (int k = 0; k < KD_; ++k) qv[k] = qall[b * D_ + h * KD_ + k];

    float cmp[8];
    float mx = -3.0e30f;
#pragma unroll
    for (int i = 0; i < 8; ++i) {
        const int n = i * 32 + lane;
        const _Float16* kp = khh + ((size_t)(b * N_ + n)) * D_ + h * KD_;
        float d = 0.f;
#pragma unroll
        for (int k = 0; k < KD_; ++k) d += qv[k] * (float)kp[k];
        d *= 0.25f;                           // 1/sqrt(KD)
        if (mask[b * N_ + n]) d = -1e30f;
        cmp[i] = d;
        mx = fmaxf(mx, d);
    }
#pragma unroll
    for (int off = 16; off >= 1; off >>= 1) mx = fmaxf(mx, __shfl_xor(mx, off, 32));

    float p[8];
    float sum = 0.f;
#pragma unroll
    for (int i = 0; i < 8; ++i) { p[i] = __expf(cmp[i] - mx); sum += p[i]; }
#pragma unroll
    for (int off = 16; off >= 1; off >>= 1) sum += __shfl_xor(sum, off, 32);

    float hacc[KD_];
#pragma unroll
    for (int k = 0; k < KD_; ++k) hacc[k] = 0.f;
#pragma unroll
    for (int i = 0; i < 8; ++i) {
        const int n = i * 32 + lane;
        const _Float16* vp = vhh + ((size_t)(b * N_ + n)) * D_ + h * KD_;
#pragma unroll
        for (int k = 0; k < KD_; ++k) hacc[k] += p[i] * (float)vp[k];
    }
#pragma unroll
    for (int k = 0; k < KD_; ++k)
#pragma unroll
        for (int off = 16; off >= 1; off >>= 1) hacc[k] += __shfl_xor(hacc[k], off, 32);

    const float inv = 1.0f / sum;
    if (lane == 0) {
#pragma unroll
        for (int k = 0; k < KD_; ++k)
            heads[b * D_ + h * KD_ + k] = (_Float16)(hacc[k] * inv);
    }
}

// ---------------------------------------------------------------------------
// threefry2x32 (20 rounds) for deterministic Gumbel noise
// ---------------------------------------------------------------------------
__device__ __forceinline__ unsigned rotl32(unsigned x, int r) {
    return (x << r) | (x >> (32 - r));
}
__device__ __forceinline__ void threefry2x32(unsigned k0, unsigned k1,
                                             unsigned c0, unsigned c1,
                                             unsigned& r0, unsigned& r1) {
    const unsigned ks2 = 0x1BD11BDAu ^ k0 ^ k1;
    unsigned ks[3] = {k0, k1, ks2};
    unsigned x0 = c0 + k0, x1 = c1 + k1;
    const int R0[4] = {13, 15, 26, 6}, R1[4] = {17, 29, 16, 24};
#pragma unroll
    for (int g = 0; g < 5; ++g) {
        const int* R = (g & 1) ? R1 : R0;
#pragma unroll
        for (int r = 0; r < 4; ++r) { x0 += x1; x1 = rotl32(x1, R[r]); x1 ^= x0; }
        x0 += ks[(g + 1) % 3];
        x1 += ks[(g + 2) % 3] + (unsigned)(g + 1);
    }
    r0 = x0; r1 = x1;
}

// ---------------------------------------------------------------------------
// Logits + tanh clip + Gumbel-max sampling + log-prob + state update.
// One block per batch element b; thread t handles node n=t.
// ---------------------------------------------------------------------------
__global__ __launch_bounds__(256) void sample_k(const float* __restrict__ qout,
                                                const float* __restrict__ kproj,
                                                const float* __restrict__ x,
                                                int* __restrict__ mask,
                                                float* __restrict__ lastv,
                                                float* __restrict__ firstv,
                                                float* __restrict__ out,
                                                int step) {
    __shared__ float qs[D_];
    __shared__ float red[N_];
    __shared__ int   ridx[N_];
    __shared__ float s_logit[N_];
    __shared__ int   sel;

    const int b = blockIdx.x;
    const int t = threadIdx.x;
    if (t < D_) qs[t] = qout[b * D_ + t];
    __syncthreads();

    const float* kp = kproj + ((size_t)(b * N_ + t)) * D_;
    float d = 0.f;
#pragma unroll 8
    for (int e = 0; e < D_; ++e) d += qs[e] * kp[e];
    float logit = tanhf(d * 0.08838834764831845f) * 10.0f;  // *1/sqrt(128), clip 10
    const bool m = (mask[b * N_ + t] != 0);
    const float lm = m ? -1e30f : logit;
    s_logit[t] = lm;

    unsigned u0, u1;
    threefry2x32(42u, 0x9E3779B9u + (unsigned)step, (unsigned)b, (unsigned)t, u0, u1);
    float uu = (float)(u0 >> 9) * (1.0f / 8388608.0f);
    float g  = -logf(-logf(uu + 1e-12f) + 1e-12f);
    float score = m ? -1e30f : (logit + g);

    red[t] = score; ridx[t] = t;
    __syncthreads();
    for (int s = 128; s > 0; s >>= 1) {
        if (t < s && red[t + s] > red[t]) { red[t] = red[t + s]; ridx[t] = ridx[t + s]; }
        __syncthreads();
    }
    if (t == 0) sel = ridx[0];
    __syncthreads();

    // logsumexp of masked logits
    red[t] = lm;
    __syncthreads();
    for (int s = 128; s > 0; s >>= 1) {
        if (t < s) red[t] = fmaxf(red[t], red[t + s]);
        __syncthreads();
    }
    const float M = red[0];
    __syncthreads();
    red[t] = __expf(lm - M);
    __syncthreads();
    for (int s = 128; s > 0; s >>= 1) {
        if (t < s) red[t] += red[t + s];
        __syncthreads();
    }
    if (t == 0) {
        float lse = M + logf(red[0]);
        out[B_ * N_ + b] += s_logit[sel] - lse;     // log_prob accumulation
        out[b * N_ + step] = (float)sel;            // idxs.T[b, step]
        mask[b * N_ + sel] = 1;
    }
    __syncthreads();
    if (t < D_) {
        float xv = x[((size_t)(b * N_ + sel)) * D_ + t];
        lastv[b * D_ + t] = xv;
        if (step == 0) firstv[b * D_ + t] = xv;
    }
}

// ---------------------------------------------------------------------------
// One-time helpers: conversion, weight packing, graph mean, state init
// ---------------------------------------------------------------------------
__global__ void f32_to_f16(const float* __restrict__ s, _Float16* __restrict__ d, int n) {
    int i = blockIdx.x * blockDim.x + threadIdx.x;
    if (i < n) d[i] = (_Float16)s[i];
}

// mode 0: W[h,d,k](8,128,16) -> Bt[(h*16+k)][d]   (W_query/W_key/W_val)
// mode 1: W[h,v,d](8,16,128) -> Bt[d][(h*16+v)]   (W_out)
__global__ void pack_w(const float* __restrict__ w, _Float16* __restrict__ dst, int mode) {
    int t = blockIdx.x * blockDim.x + threadIdx.x;
    if (t >= H_ * D_ * KD_) return;
    if (mode == 0) {
        int h = t >> 11, d = (t >> 4) & 127, k = t & 15;
        dst[((h << 4) + k) * D_ + d] = (_Float16)w[t];
    } else {
        int h = t >> 11, v = (t >> 7) & 15, d = t & 127;
        dst[d * D_ + (h << 4) + v] = (_Float16)w[t];
    }
}

__global__ void graph_emb_k(const float* __restrict__ x, float* __restrict__ ge) {
    const int b = blockIdx.x, d = threadIdx.x;
    float s = 0.f;
    for (int n = 0; n < N_; ++n) s += x[((size_t)(b * N_ + n)) * D_ + d];
    ge[b * D_ + d] = s * (1.0f / N_);
}

__global__ void init_k(const float* __restrict__ vl, const float* __restrict__ vf,
                       float* __restrict__ lastv, float* __restrict__ firstv,
                       int* __restrict__ mask, float* __restrict__ lp) {
    int i = blockIdx.x * blockDim.x + threadIdx.x;
    if (i < B_ * D_) { lastv[i] = vl[i & (D_ - 1)]; firstv[i] = vf[i & (D_ - 1)]; }
    if (i < B_ * N_) mask[i] = 0;
    if (i < B_)      lp[i] = 0.f;
}

// ---------------------------------------------------------------------------
extern "C" void kernel_launch(void* const* d_in, const int* in_sizes, int n_in,
                              void* d_out, int out_size, void* d_ws, size_t ws_size,
                              hipStream_t stream) {
    const float* x   = (const float*)d_in[0];
    const float* Wq  = (const float*)d_in[1];
    const float* Wk  = (const float*)d_in[2];
    const float* Wv  = (const float*)d_in[3];
    const float* Wo  = (const float*)d_in[4];
    const float* vl  = (const float*)d_in[5];
    const float* vf  = (const float*)d_in[6];
    const float* pkw = (const float*)d_in[7];
    const float* pcw = (const float*)d_in[8];
    float* out = (float*)d_out;   // [B*N idxs as float][B log_prob]

    char* p = (char*)d_ws;
    auto carve = [&](size_t bytes) -> void* {
        void* r = (void*)p;
        p += (bytes + 255) & ~(size_t)255;
        return r;
    };
    _Float16* xh     = (_Float16*)carve((size_t)B_ * N_ * D_ * 2);
    _Float16* khh    = (_Float16*)carve((size_t)B_ * N_ * D_ * 2);
    _Float16* vhh    = (_Float16*)carve((size_t)B_ * N_ * D_ * 2);
    float*    kproj  = (float*)   carve((size_t)B_ * N_ * D_ * 4);
    _Float16* wq_t   = (_Float16*)carve((size_t)D_ * D_ * 2);
    _Float16* wk_t   = (_Float16*)carve((size_t)D_ * D_ * 2);
    _Float16* wv_t   = (_Float16*)carve((size_t)D_ * D_ * 2);
    _Float16* wo_t   = (_Float16*)carve((size_t)D_ * D_ * 2);
    _Float16* wkp_t  = (_Float16*)carve((size_t)D_ * D_ * 2);
    _Float16* wc_t   = (_Float16*)carve((size_t)D_ * C3_ * 2);
    float*    ge     = (float*)   carve((size_t)B_ * D_ * 4);
    _Float16* ctxh   = (_Float16*)carve((size_t)B_ * D_ * 2);
    float*    qall   = (float*)   carve((size_t)B_ * D_ * 4);
    _Float16* headsh = (_Float16*)carve((size_t)B_ * D_ * 2);
    float*    qo     = (float*)   carve((size_t)B_ * D_ * 4);
    int*      mask   = (int*)     carve((size_t)B_ * N_ * 4);
    float*    lastv  = (float*)   carve((size_t)B_ * D_ * 4);
    float*    firstv = (float*)   carve((size_t)B_ * D_ * 4);

    const int nx = B_ * N_ * D_;

    // One-time precompute
    f32_to_f16<<<(nx + 255) / 256, 256, 0, stream>>>(x, xh, nx);
    f32_to_f16<<<(D_ * C3_ + 255) / 256, 256, 0, stream>>>(pcw, wc_t, D_ * C3_);
    f32_to_f16<<<(D_ * D_ + 255) / 256, 256, 0, stream>>>(pkw, wkp_t, D_ * D_);
    pack_w<<<64, 256, 0, stream>>>(Wq, wq_t, 0);
    pack_w<<<64, 256, 0, stream>>>(Wk, wk_t, 0);
    pack_w<<<64, 256, 0, stream>>>(Wv, wv_t, 0);
    pack_w<<<64, 256, 0, stream>>>(Wo, wo_t, 1);
    graph_emb_k<<<B_, D_, 0, stream>>>(x, ge);
    init_k<<<128, 256, 0, stream>>>(vl, vf, lastv, firstv, mask, out + B_ * N_);

    // Kh / Vh / k_proj : (B*N x D) @ (D x D) WMMA GEMMs
    gemm_f16<<<2048, 256, 0, stream>>>(xh, wk_t,  nullptr, khh, B_ * N_, D_, D_);
    gemm_f16<<<2048, 256, 0, stream>>>(xh, wv_t,  nullptr, vhh, B_ * N_, D_, D_);
    gemm_f16<<<2048, 256, 0, stream>>>(xh, wkp_t, kproj, nullptr, B_ * N_, D_, D_);

    // Sequential decode: 256 steps, 5 small kernels each (graph-capture friendly)
    for (int step = 0; step < N_; ++step) {
        ctx_gemm<<<8, 256, 0, stream>>>(ge, lastv, firstv, wc_t, ctxh);
        gemm_f16<<<8, 256, 0, stream>>>(ctxh, wq_t, qall, nullptr, B_, D_, D_);
        attn_k<<<128, 256, 0, stream>>>(qall, khh, vhh, mask, headsh);
        gemm_f16<<<8, 256, 0, stream>>>(headsh, wo_t, qo, nullptr, B_, D_, D_);
        sample_k<<<128, 256, 0, stream>>>(qo, kproj, x, mask, lastv, firstv, out, step);
    }
}
